// GAGNN_homo_81544249082502
// MI455X (gfx1250) — compile-verified, matrix operand order checked by reading, not run on hardware
//
#include <hip/hip_runtime.h>

#define PI_F 3.14159265358979323846f

typedef __attribute__((ext_vector_type(16))) _Float16     v16h;
typedef __attribute__((ext_vector_type(8)))  float        v8f;
typedef __attribute__((ext_vector_type(4)))  unsigned int v4u;

// Load one 16x32 f16 WMMA fragment (A or B) given the per-lane base pointer.
// Per CDNA5 ISA layout: lanes 0-15 hold K=[0..7]+[16..23], lanes 16-31 hold
// K=[8..15]+[24..31] (caller folds the +8 half-select into the pointer).
__device__ __forceinline__ v16h ld_frag(const _Float16* p) {
    union { v4u u[2]; v16h h; } c;
    c.u[0] = *(const v4u*)(p);        // K offsets +0..7   (8 halves, 16B)
    c.u[1] = *(const v4u*)(p + 16);   // K offsets +16..23 (8 halves, 16B)
    return c.h;
}

__device__ __forceinline__ v8f wmma32(v16h a, v16h b, v8f c) {
    return __builtin_amdgcn_wmma_f32_16x16x32_f16(false, a, false, b,
                                                  (short)0, c, false, false);
}

// Fast silu: v_exp_f32 + v_rcp_f32 (avoid IEEE division expansion)
__device__ __forceinline__ float silu(float x) {
    return x * __builtin_amdgcn_rcpf(1.0f + __expf(-x));
}

// ---------------------------------------------------------------------------
// Utility kernels
// ---------------------------------------------------------------------------
__global__ void k_zero(float* __restrict__ p, int n) {
    int i = blockIdx.x * 256 + threadIdx.x;
    if (i < n) p[i] = 0.0f;
}

__global__ void k_add(float* __restrict__ dst, const float* __restrict__ src, int n) {
    int i = blockIdx.x * 256 + threadIdx.x;
    if (i < n) dst[i] += src[i];
}

// Convert fp32 weight [K][N] (row-major) -> fp16 transposed [N][K]
__global__ void k_cvt_t(const float* __restrict__ src, _Float16* __restrict__ dst,
                        int K, int N) {
    int idx = blockIdx.x * 256 + threadIdx.x;
    if (idx >= K * N) return;
    int k = idx / N, n = idx - k * N;
    dst[(size_t)n * K + k] = (_Float16)src[idx];
}

// state[n,0,:] = emb[type[n],:], other 7 slots zero
__global__ void k_init_state(float* __restrict__ state, const int* __restrict__ an,
                             const float* __restrict__ emb) {
    int n = blockIdx.x, d = threadIdx.x;
    float* s = state + (size_t)n * 1024;
    s[d] = emb[an[n] * 128 + d];
#pragma unroll
    for (int g = 1; g < 8; ++g) s[g * 128 + d] = 0.0f;
}

// ---------------------------------------------------------------------------
// Edge message kernel: 16 edges / block, 128 threads (4 waves).
//   rbf_out = (sin(r*f)/r @ rbfW + b) * cutoff            (VALU, into LDS f16)
//   h   = silu(sv0 @ W1 + b1)      GEMM1: K=128,  N=640    (WMMA)
//   phi = h @ W2 + b2              GEMM2: K=640,  N=640    (WMMA)
//   gated = phi * rbf_out ; atomic scatter into msg[recv]
// ---------------------------------------------------------------------------
__global__ void __launch_bounds__(128)
k_edge(const float* __restrict__ state, float* __restrict__ msg,
       const int* __restrict__ elist, const float* __restrict__ elen,
       const float* __restrict__ evec,
       const float* __restrict__ rbfW, const float* __restrict__ rbfB,
       const _Float16* __restrict__ W1t, const float* __restrict__ b1,
       const _Float16* __restrict__ W2t, const float* __restrict__ b2,
       int E) {
    __shared__ __align__(16) _Float16 shA[16 * 128];   // sv0 tile (f16)
    __shared__ __align__(16) _Float16 shH[16 * 640];   // hidden (f16)
    __shared__ __align__(16) _Float16 shG[16 * 640];   // rbf_out -> gated (f16)
    __shared__ float shR[16 * 20];
    __shared__ float shC[16];
    __shared__ int   shS[16], shRv[16];

    const int tid = threadIdx.x;
    const int e0  = blockIdx.x * 16;

    if (tid < 16) {
        int eg   = min(e0 + tid, E - 1);
        shS[tid] = elist[2 * eg];
        shRv[tid] = elist[2 * eg + 1];
        float el = elen[eg];
        float c  = (el < 5.0f) ? 0.5f * (__cosf(el * (PI_F / 5.0f)) + 1.0f) : 0.0f;
        shC[tid] = (e0 + tid < E) ? c : 0.0f;
    }
    for (int idx = tid; idx < 16 * 20; idx += 128) {
        int e = idx / 20, k = idx - e * 20;
        float r  = fmaxf(elen[min(e0 + e, E - 1)], 1e-6f);
        float fr = (float)(k + 1) * (PI_F / 5.0f);
        shR[idx] = __sinf(r * fr) * __builtin_amdgcn_rcpf(r);
    }
    __syncthreads();
    for (int idx = tid; idx < 16 * 640; idx += 128) {
        int e = idx / 640, j = idx - e * 640;
        float acc = rbfB[j];
#pragma unroll
        for (int k = 0; k < 20; ++k) acc += shR[e * 20 + k] * rbfW[k * 640 + j];
        shG[idx] = (_Float16)(acc * shC[e]);
    }
    for (int idx = tid; idx < 16 * 128; idx += 128) {
        int e = idx >> 7, d = idx & 127;
        shA[idx] = (_Float16)state[(size_t)shS[e] * 1024 + d];
    }
    __syncthreads();

    const int lane  = tid & 31;
    const int wave  = tid >> 5;
    const int r16   = lane & 15;
    const int khalf = (lane >> 4) << 3;      // +8 K-halves for upper lanes
    const int rbase = (lane < 16) ? 0 : 8;

    // GEMM1: 40 col-tiles -> 10 per wave, K=128 -> 4 WMMA each
    for (int ti = 0; ti < 10; ++ti) {
        int j0 = (wave * 10 + ti) * 16;
        v8f acc = {};
#pragma unroll
        for (int kc = 0; kc < 4; ++kc) {
            int k0 = kc * 32;
            v16h a = ld_frag(shA + r16 * 128 + k0 + khalf);
            v16h b = ld_frag(W1t + (size_t)(j0 + r16) * 128 + k0 + khalf);
            acc = wmma32(a, b, acc);
        }
        int col = j0 + r16;
        float bias = b1[col];
#pragma unroll
        for (int i = 0; i < 8; ++i)
            shH[(rbase + i) * 640 + col] = (_Float16)silu(acc[i] + bias);
    }
    __syncthreads();

    // GEMM2: K=640 -> 20 WMMA per tile; gated written in place over rbf_out
    for (int ti = 0; ti < 10; ++ti) {
        int j0 = (wave * 10 + ti) * 16;
        v8f acc = {};
#pragma unroll
        for (int kc = 0; kc < 20; ++kc) {
            int k0 = kc * 32;
            __builtin_prefetch(W2t + (size_t)(j0 + r16) * 640 + k0 + 64, 0, 0);
            v16h a = ld_frag(shH + r16 * 640 + k0 + khalf);
            v16h b = ld_frag(W2t + (size_t)(j0 + r16) * 640 + k0 + khalf);
            acc = wmma32(a, b, acc);
        }
        int col = j0 + r16;
        float bias = b2[col];
#pragma unroll
        for (int i = 0; i < 8; ++i) {
            int row = rbase + i;
            float rb = (float)shG[row * 640 + col];
            shG[row * 640 + col] = (_Float16)((acc[i] + bias) * rb);
        }
    }
    __syncthreads();

    // Scatter: msg[recv] += {g_s, g_v*sv_v + g_d*ev, g_b*sv_b, g_t*sv_t}
    for (int idx = tid; idx < 16 * 128; idx += 128) {
        int e = idx >> 7, d = idx & 127;
        int eg = min(e0 + e, E - 1);
        const float* sv = state + (size_t)shS[e] * 1024;
        float* mg = msg + (size_t)shRv[e] * 1024;
        float gs = (float)shG[e * 640 + d];
        float gv = (float)shG[e * 640 + 128 + d];
        float gd = (float)shG[e * 640 + 256 + d];
        float gb = (float)shG[e * 640 + 384 + d];
        float gt = (float)shG[e * 640 + 512 + d];
        atomicAdd(mg + d, gs);
#pragma unroll
        for (int c = 0; c < 3; ++c) {
            float ev = evec[(size_t)eg * 3 + c];
            atomicAdd(mg + (1 + c) * 128 + d, gv * sv[(1 + c) * 128 + d] + gd * ev);
            atomicAdd(mg + (4 + c) * 128 + d, gb * sv[(4 + c) * 128 + d]);
        }
        atomicAdd(mg + 7 * 128 + d, gt * sv[7 * 128 + d]);
    }
}

// ---------------------------------------------------------------------------
// Row-wise GEMM: Y[64 rows/block] (+)= X @ Wt^T, K=N=128 (U, V, Z1->Z1l)
// ---------------------------------------------------------------------------
__global__ void __launch_bounds__(128)
k_gemm_rows(const float* __restrict__ X, const _Float16* __restrict__ Wt,
            float* __restrict__ Y, int accumulate) {
    __shared__ __align__(16) _Float16 shA[64 * 128];
    const int tid = threadIdx.x;
    const size_t row0 = (size_t)blockIdx.x * 64;
    for (int idx = tid; idx < 64 * 128; idx += 128)
        shA[idx] = (_Float16)X[row0 * 128 + idx];
    __syncthreads();
    const int lane = tid & 31, wave = tid >> 5;
    const int r16 = lane & 15, khalf = (lane >> 4) << 3;
    const int rbase = (lane < 16) ? 0 : 8;
    const int m0 = wave * 16;
    for (int tj = 0; tj < 8; ++tj) {
        int j0 = tj * 16;
        v8f acc = {};
#pragma unroll
        for (int kc = 0; kc < 4; ++kc) {
            int k0 = kc * 32;
            v16h a = ld_frag(shA + (m0 + r16) * 128 + k0 + khalf);
            v16h b = ld_frag(Wt + (size_t)(j0 + r16) * 128 + k0 + khalf);
            acc = wmma32(a, b, acc);
        }
        int col = j0 + r16;
#pragma unroll
        for (int i = 0; i < 8; ++i) {
            size_t o = (row0 + m0 + rbase + i) * 128 + col;
            if (accumulate) Y[o] += acc[i]; else Y[o] = acc[i];
        }
    }
}

// ---------------------------------------------------------------------------
// Geometric product: Z1 = first8 - last8 of scatter(U_a * V_b * gpw[Widx])
// ---------------------------------------------------------------------------
__global__ void k_wgp(const float* __restrict__ U, const float* __restrict__ V,
                      const float* __restrict__ gpw, float* __restrict__ Z1) {
    constexpr int GP[64] = {
        0,1,2,3,4,5,6,7,    1,0,4,14,2,7,11,5,  2,12,0,5,9,3,7,6,
        3,6,13,0,7,10,1,4,  4,10,1,7,8,14,5,11, 5,7,11,2,6,8,12,9,
        6,3,7,9,13,4,8,10,  7,5,6,4,11,9,10,8};
    constexpr int WI[64] = {
        0,1,1,1,2,2,2,3,     4,5,6,6,7,8,7,9,      4,6,5,6,7,7,8,9,
        4,6,6,5,8,7,7,9,     10,11,11,12,13,14,14,15, 10,12,11,11,14,13,14,15,
        10,11,12,11,14,14,13,15, 16,17,17,17,18,18,18,19};
    const int n = blockIdx.x, d = threadIdx.x;
    const size_t base = (size_t)n * 1024 + d;
    float u[8], v[8], gw[20], comb[16];
#pragma unroll
    for (int g = 0; g < 8; ++g) { u[g] = U[base + g * 128]; v[g] = V[base + g * 128]; }
#pragma unroll
    for (int w = 0; w < 20; ++w) gw[w] = gpw[w * 128 + d];
#pragma unroll
    for (int i = 0; i < 16; ++i) comb[i] = 0.0f;
#pragma unroll
    for (int a = 0; a < 8; ++a)
#pragma unroll
        for (int b = 0; b < 8; ++b)
            comb[GP[a * 8 + b]] += u[a] * v[b] * gw[WI[a * 8 + b]];
#pragma unroll
    for (int g = 0; g < 8; ++g) Z1[base + g * 128] = comb[g] - comb[8 + g];
}

// upd_in = [state[:,0], ||V[:,1:4]||]
__global__ void k_updin(const float* __restrict__ state, const float* __restrict__ V,
                        float* __restrict__ updin) {
    int n = blockIdx.x, d = threadIdx.x;
    size_t b = (size_t)n * 1024 + d;
    float v1 = V[b + 128], v2 = V[b + 256], v3 = V[b + 384];
    updin[(size_t)n * 256 + d] = state[b];
    updin[(size_t)n * 256 + 128 + d] =
        __builtin_amdgcn_sqrtf(v1 * v1 + v2 * v2 + v3 * v3 + 1e-12f);
}

// ---------------------------------------------------------------------------
// Per-type update MLP: a[n] = silu(updin @ W1 + b1) @ W2 + b2, masked by type.
// grid.z = type; tile of 16 nodes; early exit when no row has this type.
// ---------------------------------------------------------------------------
__global__ void __launch_bounds__(128)
k_upd_mlp(const float* __restrict__ updin, const int* __restrict__ an,
          const _Float16* __restrict__ W1t, const float* __restrict__ b1,
          const _Float16* __restrict__ W2t, const float* __restrict__ b2,
          float* __restrict__ a) {
    __shared__ __align__(16) _Float16 shA[16 * 256];
    __shared__ __align__(16) _Float16 shH[16 * 512];
    __shared__ int shT[16];
    __shared__ int shFlag;
    const int t = blockIdx.z;
    const int n0 = blockIdx.x * 16;
    const int tid = threadIdx.x;
    if (tid == 0) shFlag = 0;
    __syncthreads();
    if (tid < 16) { int ty = an[n0 + tid]; shT[tid] = ty; if (ty == t) shFlag = 1; }
    __syncthreads();
    if (!shFlag) return;
    for (int idx = tid; idx < 16 * 256; idx += 128)
        shA[idx] = (_Float16)updin[(size_t)n0 * 256 + idx];
    __syncthreads();
    const _Float16* w1 = W1t + (size_t)t * 512 * 256;
    const _Float16* w2 = W2t + (size_t)t * 512 * 512;
    const float* bb1 = b1 + t * 512;
    const float* bb2 = b2 + t * 512;
    const int lane = tid & 31, wave = tid >> 5;
    const int r16 = lane & 15, khalf = (lane >> 4) << 3;
    const int rbase = (lane < 16) ? 0 : 8;
    for (int tj = 0; tj < 8; ++tj) {               // hidden 512 = 32 tiles / 4 waves
        int j0 = (wave * 8 + tj) * 16;
        v8f acc = {};
#pragma unroll
        for (int kc = 0; kc < 8; ++kc) {           // K = 256
            int k0 = kc * 32;
            v16h av = ld_frag(shA + r16 * 256 + k0 + khalf);
            v16h bv = ld_frag(w1 + (size_t)(j0 + r16) * 256 + k0 + khalf);
            acc = wmma32(av, bv, acc);
        }
        int col = j0 + r16;
        float bias = bb1[col];
#pragma unroll
        for (int i = 0; i < 8; ++i)
            shH[(rbase + i) * 512 + col] = (_Float16)silu(acc[i] + bias);
    }
    __syncthreads();
    for (int tj = 0; tj < 8; ++tj) {               // out 512, K = 512
        int j0 = (wave * 8 + tj) * 16;
        v8f acc = {};
#pragma unroll
        for (int kc = 0; kc < 16; ++kc) {
            int k0 = kc * 32;
            v16h av = ld_frag(shH + r16 * 512 + k0 + khalf);
            v16h bv = ld_frag(w2 + (size_t)(j0 + r16) * 512 + k0 + khalf);
            acc = wmma32(av, bv, acc);
        }
        int col = j0 + r16;
        float bias = bb2[col];
#pragma unroll
        for (int i = 0; i < 8; ++i) {
            int row = rbase + i;
            if (shT[row] == t)
                a[(size_t)(n0 + row) * 512 + col] = acc[i] + bias;
        }
    }
}

// state += a_chunk * UZ per slot group
__global__ void k_state_update(float* __restrict__ state, const float* __restrict__ UZ,
                               const float* __restrict__ a) {
    int n = blockIdx.x, d = threadIdx.x;
    size_t sb = (size_t)n * 1024 + d;
    const float* an_ = a + (size_t)n * 512;
    float as = an_[d], av = an_[128 + d], ab = an_[256 + d], at_ = an_[384 + d];
    state[sb] += as * UZ[sb];
#pragma unroll
    for (int c = 1; c < 4; ++c) state[sb + c * 128] += av * UZ[sb + c * 128];
#pragma unroll
    for (int c = 4; c < 7; ++c) state[sb + c * 128] += ab * UZ[sb + c * 128];
    state[sb + 7 * 128] += at_ * UZ[sb + 7 * 128];
}

// readout: out[graph[n]] += silu(state[n,0] @ roW1 + b1) @ roW2 + b2
__global__ void __launch_bounds__(64)
k_readout(const float* __restrict__ state, const int* __restrict__ gi,
          const float* __restrict__ W1, const float* __restrict__ b1,
          const float* __restrict__ W2, const float* __restrict__ b2,
          float* __restrict__ out) {
    __shared__ float sh[64];
    int n = blockIdx.x, j = threadIdx.x;
    const float* s = state + (size_t)n * 1024;
    float acc = b1[j];
    for (int k = 0; k < 128; ++k) acc += s[k] * W1[k * 64 + j];
    sh[j] = silu(acc) * W2[j];
    __syncthreads();
    if (j == 0) {
        float c = b2[0];
        for (int k = 0; k < 64; ++k) c += sh[k];
        atomicAdd(&out[gi[n]], c);
    }
}

// ---------------------------------------------------------------------------
extern "C" void kernel_launch(void* const* d_in, const int* in_sizes, int n_in,
                              void* d_out, int out_size, void* d_ws, size_t ws_size,
                              hipStream_t stream) {
    const int* an    = (const int*)d_in[0];
    const int* elist = (const int*)d_in[1];
    const float* elen = (const float*)d_in[2];
    const float* evec = (const float*)d_in[3];
    const int* gidx  = (const int*)d_in[4];
    const float* emb  = (const float*)d_in[5];
    const float* rbfW = (const float*)d_in[6];
    const float* rbfB = (const float*)d_in[7];
    const float* phiW1 = (const float*)d_in[8];
    const float* phiB1 = (const float*)d_in[9];
    const float* phiW2 = (const float*)d_in[10];
    const float* phiB2 = (const float*)d_in[11];
    const float* UW   = (const float*)d_in[12];
    const float* VW   = (const float*)d_in[13];
    const float* gpw  = (const float*)d_in[14];
    const float* z1W  = (const float*)d_in[15];
    const float* upW1 = (const float*)d_in[16];
    const float* upB1 = (const float*)d_in[17];
    const float* upW2 = (const float*)d_in[18];
    const float* upB2 = (const float*)d_in[19];
    const float* roW1 = (const float*)d_in[20];
    const float* roB1 = (const float*)d_in[21];
    const float* roW2 = (const float*)d_in[22];
    const float* roB2 = (const float*)d_in[23];

    const int N = in_sizes[0];
    const int E = in_sizes[2];
    const int G = out_size;

    char* wp = (char*)d_ws;
    auto carve = [&](size_t bytes) {
        void* q = (void*)wp;
        wp += (bytes + 255) & ~(size_t)255;
        return q;
    };
    float* state = (float*)carve((size_t)N * 1024 * 4);
    float* msg   = (float*)carve((size_t)N * 1024 * 4);   // msg buf, then Z1
    float* U     = (float*)carve((size_t)N * 1024 * 4);   // U, then UZ = U + Z1l
    float* V     = (float*)carve((size_t)N * 1024 * 4);
    float* updin = (float*)carve((size_t)N * 256 * 4);
    float* abuf  = (float*)carve((size_t)N * 512 * 4);
    _Float16* pW1t  = (_Float16*)carve((size_t)640 * 128 * 2);
    _Float16* pW2t  = (_Float16*)carve((size_t)640 * 640 * 2);
    _Float16* pUt   = (_Float16*)carve((size_t)128 * 128 * 2);
    _Float16* pVt   = (_Float16*)carve((size_t)128 * 128 * 2);
    _Float16* pZt   = (_Float16*)carve((size_t)128 * 128 * 2);
    _Float16* pUp1t = (_Float16*)carve((size_t)5 * 512 * 256 * 2);
    _Float16* pUp2t = (_Float16*)carve((size_t)5 * 512 * 512 * 2);
    (void)ws_size; (void)n_in;

    auto cvt = [&](const float* src, _Float16* dst, int K, int Nn) {
        int tot = K * Nn;
        k_cvt_t<<<(tot + 255) / 256, 256, 0, stream>>>(src, dst, K, Nn);
    };

    k_init_state<<<N, 128, 0, stream>>>(state, an, emb);

    const int stElems = N * 1024;
    for (int r = 0; r < 4; ++r) {
        // per-round weight prep (fp16, transposed)
        cvt(phiW1 + (size_t)r * 128 * 640, pW1t, 128, 640);
        cvt(phiW2 + (size_t)r * 640 * 640, pW2t, 640, 640);
        cvt(UW + (size_t)r * 128 * 128, pUt, 128, 128);
        cvt(VW + (size_t)r * 128 * 128, pVt, 128, 128);
        cvt(z1W + (size_t)r * 128 * 128, pZt, 128, 128);
        for (int t = 0; t < 5; ++t) {
            cvt(upW1 + ((size_t)r * 5 + t) * 256 * 512, pUp1t + (size_t)t * 512 * 256, 256, 512);
            cvt(upW2 + ((size_t)r * 5 + t) * 512 * 512, pUp2t + (size_t)t * 512 * 512, 512, 512);
        }

        k_zero<<<(stElems + 255) / 256, 256, 0, stream>>>(msg, stElems);
        k_edge<<<(E + 15) / 16, 128, 0, stream>>>(
            state, msg, elist, elen, evec,
            rbfW + (size_t)r * 20 * 640, rbfB + (size_t)r * 640,
            pW1t, phiB1 + (size_t)r * 640, pW2t, phiB2 + (size_t)r * 640, E);
        k_add<<<(stElems + 255) / 256, 256, 0, stream>>>(state, msg, stElems);

        const int rowBlocks = (N * 8) / 64;
        k_gemm_rows<<<rowBlocks, 128, 0, stream>>>(state, pUt, U, 0);
        k_gemm_rows<<<rowBlocks, 128, 0, stream>>>(state, pVt, V, 0);
        k_wgp<<<N, 128, 0, stream>>>(U, V, gpw + (size_t)r * 20 * 128, msg); // Z1 in msg
        k_gemm_rows<<<rowBlocks, 128, 0, stream>>>(msg, pZt, U, 1);          // U := UZ

        k_updin<<<N, 128, 0, stream>>>(state, V, updin);
        k_zero<<<(N * 512 + 255) / 256, 256, 0, stream>>>(abuf, N * 512);
        k_upd_mlp<<<dim3(N / 16, 1, 5), 128, 0, stream>>>(
            updin, an, pUp1t, upB1 + (size_t)r * 5 * 512,
            pUp2t, upB2 + (size_t)r * 5 * 512, abuf);
        k_state_update<<<N, 128, 0, stream>>>(state, U, abuf);
    }

    k_zero<<<(G + 255) / 256, 256, 0, stream>>>((float*)d_out, G);
    k_readout<<<N, 64, 0, stream>>>(state, gidx, roW1, roB1, roW2, roB2, (float*)d_out);
}
